// GKLayer_1675037245696
// MI455X (gfx1250) — compile-verified, matrix operand order checked by reading, not run on hardware
//
#include <hip/hip_runtime.h>
#include <hip/hip_bf16.h>
#include <math.h>

typedef __attribute__((ext_vector_type(16))) _Float16 v16h;
typedef __attribute__((ext_vector_type(8)))  _Float16 v8h;
typedef __attribute__((ext_vector_type(4)))  _Float16 v4h;
typedef __attribute__((ext_vector_type(8)))  float    v8f;

#define CC      64
#define FIN     134
#define FINPAD  160
#define NFRAG_L (FINPAD / 32 * 4)   // 20 B-fragments for the lift layer
#define WPB     4                    // waves per block
#define TPB     (WPB * 32)

// ---------------- WMMA helpers (CDNA5 16x16x32 f16 -> f32) ----------------

__device__ __forceinline__ int kmap(int lane, int j) {
  // 16-bit A/B operand K index for element j of this lane (ISA 7.12.2 layout)
  return ((j & 8) << 1) | ((lane & 16) >> 1) | (j & 7);
}

// 16 contiguous halves (32B, 16B-aligned) -> fragment; 2x ds_load_b128
__device__ __forceinline__ v16h load_frag_contig(const _Float16* p) {
  union { v16h v; v8h h[2]; } u;
  u.h[0] = ((const v8h*)p)[0];
  u.h[1] = ((const v8h*)p)[1];
  return u.v;
}

// A fragment from row-major LDS tile: two contiguous 8-half runs per lane
__device__ __forceinline__ v16h load_a_frag(const _Float16* tile, int lane, int ldk, int k0) {
  int m = lane & 15;
  const _Float16* p = tile + m * ldk + k0 + ((lane & 16) >> 1);
  union { v16h v; v8h h[2]; } u;
  u.h[0] = *(const v8h*)p;
  u.h[1] = *(const v8h*)(p + 16);
  return u.v;
}

__device__ __forceinline__ v8f wmma_f16(v16h a, v16h b, v8f c) {
  return __builtin_amdgcn_wmma_f32_16x16x32_f16(false, a, false, b, (short)0, c, false, false);
}

__device__ __forceinline__ v8f zero8() {
  v8f z;
#pragma unroll
  for (int i = 0; i < 8; ++i) z[i] = 0.f;
  return z;
}

__device__ __forceinline__ v4h cvt4(float4 v) {
  v4h r;
  r[0] = (_Float16)v.x; r[1] = (_Float16)v.y;
  r[2] = (_Float16)v.z; r[3] = (_Float16)v.w;
  return r;
}

// intra-wave LDS producer->consumer ordering (DS ops are in-order per wave)
__device__ __forceinline__ void wave_lds_sync() {
  __builtin_amdgcn_wave_barrier();
  asm volatile("" ::: "memory");
}

__device__ __forceinline__ float elu1(float v) { return v > 0.f ? v : (expf(v) - 1.f); }

// fragment-major fill of a 64x64 weight: 8 frags x [32 lanes][16 halves]
__device__ __forceinline__ void fill_wfrag64(_Float16* sWf, const float* __restrict__ W, int tid) {
  for (int idx = tid; idx < 8 * 512; idx += TPB) {
    int f = idx >> 9, l = (idx >> 4) & 31, j = idx & 15;
    int k = ((f >> 2) << 5) + kmap(l, j);
    int c = ((f & 3) << 4) + (l & 15);
    sWf[idx] = (_Float16)W[k * CC + c];
  }
}

// per-column BN affine table: sbn[c] = g*inv, sbn[CC+c] = b - mu*g*inv
__device__ __forceinline__ void fill_bn(float* sbn, const float* __restrict__ mu_inv,
                                        const float* __restrict__ g, const float* __restrict__ b,
                                        int tid) {
  if (tid < CC) {
    float sc = g[tid] * mu_inv[CC + tid];
    sbn[tid] = sc;
    sbn[CC + tid] = b[tid] - mu_inv[tid] * sc;
  }
}

// permuted-K W_lift element: K layout = [y_src 0..63 | y_dst 64..127 | pos_s 128..130 | pos_d 131..133 | pad]
__device__ __forceinline__ float wlift_elem(const float* __restrict__ W, int k, int c) {
  int r;
  if (k < 64)       r = 6 + k;
  else if (k < 128) r = 70 + (k - 64);
  else if (k < 131) r = k - 128;
  else if (k < 134) r = 3 + (k - 131);
  else return 0.f;
  return W[r * CC + c];
}

// ---------------- kernels ----------------

__global__ void k_init(float* deg, float* agg, float* stats, int N) {
  int i = blockIdx.x * blockDim.x + threadIdx.x;
  if (i < N * CC) agg[i] = 0.f;
  if (i < N)      deg[i] = 1.f;       // self loop contributes 1 to out-degree
  if (i < 6 * CC) stats[i] = 0.f;
}

__global__ void k_degree(const int* __restrict__ ei, float* deg, int E) {
  int e = blockIdx.x * blockDim.x + threadIdx.x;
  if (e < E) atomicAdd(&deg[ei[e]], 1.f);     // ei row 0 = src
}

// h = feat @ W_lift + b_lift (K-permuted), accumulate BN stats
__global__ __launch_bounds__(TPB) void k_edge_lift(
    const float* __restrict__ pos, const float* __restrict__ y,
    const int* __restrict__ ei, const float* __restrict__ W,
    const float* __restrict__ bias, _Float16* __restrict__ h,
    float* __restrict__ stats, long E, long E2, int ntiles)
{
  __shared__ _Float16 sWf[NFRAG_L * 512];        // 20KB fragment-major weights
  __shared__ _Float16 sb[CC];
  __shared__ _Float16 sfeat[WPB][16 * FINPAD];   // 20KB
  __shared__ int ssrc[WPB][16];
  __shared__ int sdst[WPB][16];

  int tid = threadIdx.x;
  for (int idx = tid; idx < NFRAG_L * 512; idx += TPB) {
    int f = idx >> 9, l = (idx >> 4) & 31, j = idx & 15;
    int k = ((f >> 2) << 5) + kmap(l, j);
    int c = ((f & 3) << 4) + (l & 15);
    sWf[idx] = (_Float16)wlift_elem(W, k, c);
  }
  if (tid < CC) sb[tid] = (_Float16)bias[tid];
  __syncthreads();

  int wave = tid >> 5, lane = tid & 31;
  int gw = blockIdx.x * WPB + wave, stride = gridDim.x * WPB;

  for (int t = gw; t < ntiles; t += stride) {
    long base = (long)t * 16;
    if (lane < 16) {
      long e = base + lane;
      int s, d;
      if (e < E)        { s = ei[e]; d = ei[E + e]; }
      else if (e < E2)  { s = d = (int)(e - E); }          // self loop
      else              { s = -1; d = 0; }                 // pad row
      ssrc[wave][lane] = s;
      sdst[wave][lane] = d;
    }
    wave_lds_sync();

    // y region K=0..127: float4 gather -> packed 4xf16 LDS stores
    for (int i = lane; i < 16 * 32; i += 32) {
      int r = i >> 5, q = i & 31;
      int s = ssrc[wave][r];
      int node = (q < 16) ? s : sdst[wave][r];
      float4 v = {0.f, 0.f, 0.f, 0.f};
      if (s >= 0) v = ((const float4*)y)[(long)node * 16 + (q & 15)];
      *(v4h*)(sfeat[wave] + r * FINPAD + q * 4) = cvt4(v);
    }
    // pos + pad region K=128..159
    for (int i = lane; i < 16 * 32; i += 32) {
      int r = i >> 5, kk = i & 31;
      int s = ssrc[wave][r], d = sdst[wave][r];
      float v = 0.f;
      if (s >= 0 && kk < 6) v = (kk < 3) ? pos[(long)s * 3 + kk] : pos[(long)d * 3 + (kk - 3)];
      sfeat[wave][r * FINPAD + 128 + kk] = (_Float16)v;
    }
    wave_lds_sync();

    v8f acc[4];
#pragma unroll
    for (int nc = 0; nc < 4; ++nc) acc[nc] = zero8();

#pragma unroll
    for (int kc = 0; kc < FINPAD / 32; ++kc) {
      v16h a = load_a_frag(sfeat[wave], lane, FINPAD, kc * 32);
#pragma unroll
      for (int nc = 0; nc < 4; ++nc) {
        v16h b = load_frag_contig(sWf + (kc * 4 + nc) * 512 + lane * 16);
        acc[nc] = wmma_f16(a, b, acc[nc]);
      }
    }

    int n0 = lane & 15, mb = (lane >> 4) << 3;
#pragma unroll
    for (int nc = 0; nc < 4; ++nc) {
      int c = nc * 16 + n0;
      float bv = (float)sb[c];
      float s = 0.f, sq = 0.f;
#pragma unroll
      for (int v = 0; v < 8; ++v) {
        int m = mb + v;
        if (ssrc[wave][m] >= 0) {
          float val = acc[nc][v] + bv;
          h[(base + m) * CC + c] = (_Float16)val;
          s += val; sq += val * val;
        }
      }
      atomicAdd(&stats[c], s);
      atomicAdd(&stats[CC + c], sq);
    }
  }
}

__global__ void k_stats_fin(const float* __restrict__ sums, float* __restrict__ mu_inv, float count) {
  int c = threadIdx.x;
  if (c < CC) {
    float mu = sums[c] / count;
    float var = sums[CC + c] / count - mu * mu;
    mu_inv[c] = mu;
    mu_inv[CC + c] = rsqrtf(var + 1e-5f);
  }
}

// in-place h = elu(bn(h)) @ W_hid + b_hid; BN0+ELU fused into tile staging.
__global__ __launch_bounds__(TPB) void k_edge_hid(
    _Float16* __restrict__ h, const float* __restrict__ W, const float* __restrict__ bias,
    const float* __restrict__ mu_inv, const float* __restrict__ g, const float* __restrict__ bb,
    float* __restrict__ stats, long E2, int ntiles)
{
  __shared__ _Float16 sWf[8 * 512];
  __shared__ _Float16 sb[CC];
  __shared__ float    sbn[2 * CC];
  __shared__ _Float16 sA[WPB][16 * CC];

  int tid = threadIdx.x;
  fill_wfrag64(sWf, W, tid);
  fill_bn(sbn, mu_inv, g, bb, tid);
  if (tid < CC) sb[tid] = (_Float16)bias[tid];
  __syncthreads();

  int wave = tid >> 5, lane = tid & 31;
  v16h bf[8];
#pragma unroll
  for (int f = 0; f < 8; ++f) bf[f] = load_frag_contig(sWf + f * 512 + lane * 16);

  int gw = blockIdx.x * WPB + wave, stride = gridDim.x * WPB;
  for (int t = gw; t < ntiles; t += stride) {
    long base = (long)t * 16;
    for (int i = lane; i < 16 * 8; i += 32) {          // 16 rows x 8 v8h chunks
      long e = base + (i >> 3);
      v8h hv;
#pragma unroll
      for (int q = 0; q < 8; ++q) hv[q] = (_Float16)0.f;
      if (e < E2) {
        v8h raw = ((const v8h*)h)[e * 8 + (i & 7)];
        int cb = (i & 7) * 8;
#pragma unroll
        for (int q = 0; q < 8; ++q) {
          float v = (float)raw[q] * sbn[cb + q] + sbn[CC + cb + q];
          hv[q] = (_Float16)elu1(v);
        }
      }
      ((v8h*)sA[wave])[i] = hv;
    }
    wave_lds_sync();

    v8f acc[4];
#pragma unroll
    for (int nc = 0; nc < 4; ++nc) acc[nc] = zero8();
#pragma unroll
    for (int kc = 0; kc < 2; ++kc) {
      v16h a = load_a_frag(sA[wave], lane, CC, kc * 32);
#pragma unroll
      for (int nc = 0; nc < 4; ++nc) acc[nc] = wmma_f16(a, bf[kc * 4 + nc], acc[nc]);
    }

    int n0 = lane & 15, mb = (lane >> 4) << 3;
#pragma unroll
    for (int nc = 0; nc < 4; ++nc) {
      int c = nc * 16 + n0;
      float bv = (float)sb[c];
      float s = 0.f, sq = 0.f;
#pragma unroll
      for (int v = 0; v < 8; ++v) {
        long e = base + mb + v;
        if (e < E2) {
          float val = acc[nc][v] + bv;
          h[e * CC + c] = (_Float16)val;
          s += val; sq += val * val;
        }
      }
      atomicAdd(&stats[c], s);
      atomicAdd(&stats[CC + c], sq);
    }
  }
}

// ew = elu(bn1(h)) @ W_low + b_low ; scatter agg[dst] += ew/deg[dst] * x[src]
__global__ __launch_bounds__(TPB) void k_edge_low_scatter(
    const _Float16* __restrict__ h, const float* __restrict__ W, const float* __restrict__ bias,
    const float* __restrict__ mu_inv, const float* __restrict__ g, const float* __restrict__ bb,
    const int* __restrict__ ei, const float* __restrict__ x, const float* __restrict__ deg,
    float* __restrict__ agg, long E, long E2, int ntiles)
{
  __shared__ _Float16 sWf[8 * 512];
  __shared__ _Float16 sb[CC];
  __shared__ float    sbn[2 * CC];
  __shared__ _Float16 sA[WPB][16 * CC];
  __shared__ int   ssrc[WPB][16];
  __shared__ int   sdst[WPB][16];
  __shared__ float sinv[WPB][16];

  int tid = threadIdx.x;
  fill_wfrag64(sWf, W, tid);
  fill_bn(sbn, mu_inv, g, bb, tid);
  if (tid < CC) sb[tid] = (_Float16)bias[tid];
  __syncthreads();

  int wave = tid >> 5, lane = tid & 31;
  v16h bf[8];
#pragma unroll
  for (int f = 0; f < 8; ++f) bf[f] = load_frag_contig(sWf + f * 512 + lane * 16);

  int gw = blockIdx.x * WPB + wave, stride = gridDim.x * WPB;
  for (int t = gw; t < ntiles; t += stride) {
    long base = (long)t * 16;
    if (lane < 16) {
      long e = base + lane;
      int s, d;
      if (e < E)       { s = ei[e]; d = ei[E + e]; }
      else if (e < E2) { s = d = (int)(e - E); }
      else             { s = -1; d = 0; }
      ssrc[wave][lane] = s;
      sdst[wave][lane] = d;
      sinv[wave][lane] = (s >= 0) ? 1.f / deg[d] : 0.f;
    }
    for (int i = lane; i < 16 * 8; i += 32) {
      long e = base + (i >> 3);
      v8h hv;
#pragma unroll
      for (int q = 0; q < 8; ++q) hv[q] = (_Float16)0.f;
      if (e < E2) {
        v8h raw = ((const v8h*)h)[e * 8 + (i & 7)];
        int cb = (i & 7) * 8;
#pragma unroll
        for (int q = 0; q < 8; ++q) {
          float v = (float)raw[q] * sbn[cb + q] + sbn[CC + cb + q];
          hv[q] = (_Float16)elu1(v);
        }
      }
      ((v8h*)sA[wave])[i] = hv;
    }
    wave_lds_sync();

    v8f acc[4];
#pragma unroll
    for (int nc = 0; nc < 4; ++nc) acc[nc] = zero8();
#pragma unroll
    for (int kc = 0; kc < 2; ++kc) {
      v16h a = load_a_frag(sA[wave], lane, CC, kc * 32);
#pragma unroll
      for (int nc = 0; nc < 4; ++nc) acc[nc] = wmma_f16(a, bf[kc * 4 + nc], acc[nc]);
    }

    int n0 = lane & 15, mb = (lane >> 4) << 3;
#pragma unroll
    for (int nc = 0; nc < 4; ++nc) {
      int c = nc * 16 + n0;
      float bv = (float)sb[c];
#pragma unroll
      for (int v = 0; v < 8; ++v) {
        int m = mb + v;
        int s = ssrc[wave][m];
        if (s >= 0) {
          float val = (acc[nc][v] + bv) * sinv[wave][m] * x[(long)s * CC + c];
          atomicAdd(&agg[(long)sdst[wave][m] * CC + c], val);
        }
      }
    }
  }
}

// pre = x @ W_lin + agg, accumulate BN stats
__global__ __launch_bounds__(TPB) void k_node(
    const float* __restrict__ x, const float* __restrict__ W, const float* __restrict__ agg,
    float* __restrict__ pre, float* __restrict__ stats, int N, int ntiles)
{
  __shared__ _Float16 sWf[8 * 512];
  __shared__ _Float16 sA[WPB][16 * CC];

  int tid = threadIdx.x;
  fill_wfrag64(sWf, W, tid);
  __syncthreads();

  int wave = tid >> 5, lane = tid & 31;
  v16h bf[8];
#pragma unroll
  for (int f = 0; f < 8; ++f) bf[f] = load_frag_contig(sWf + f * 512 + lane * 16);

  int gw = blockIdx.x * WPB + wave, stride = gridDim.x * WPB;
  for (int t = gw; t < ntiles; t += stride) {
    long base = (long)t * 16;
    for (int i = lane; i < 16 * 16; i += 32) {         // 16 rows x 16 float4
      long e = base + (i >> 4);
      float4 v = {0.f, 0.f, 0.f, 0.f};
      if (e < N) v = ((const float4*)x)[e * 16 + (i & 15)];
      *(v4h*)(sA[wave] + (i >> 4) * CC + (i & 15) * 4) = cvt4(v);
    }
    wave_lds_sync();

    int n0 = lane & 15, mb = (lane >> 4) << 3;
    v8f acc[4];
#pragma unroll
    for (int nc = 0; nc < 4; ++nc) {
      int c = nc * 16 + n0;
#pragma unroll
      for (int v = 0; v < 8; ++v) {
        long e = base + mb + v;
        acc[nc][v] = (e < N) ? agg[e * CC + c] : 0.f;   // C := agg (free add)
      }
    }
#pragma unroll
    for (int kc = 0; kc < 2; ++kc) {
      v16h a = load_a_frag(sA[wave], lane, CC, kc * 32);
#pragma unroll
      for (int nc = 0; nc < 4; ++nc) acc[nc] = wmma_f16(a, bf[kc * 4 + nc], acc[nc]);
    }
#pragma unroll
    for (int nc = 0; nc < 4; ++nc) {
      int c = nc * 16 + n0;
      float s = 0.f, sq = 0.f;
#pragma unroll
      for (int v = 0; v < 8; ++v) {
        long e = base + mb + v;
        if (e < N) {
          float val = acc[nc][v];
          pre[e * CC + c] = val;
          s += val; sq += val * val;
        }
      }
      atomicAdd(&stats[c], s);
      atomicAdd(&stats[CC + c], sq);
    }
  }
}

__global__ void k_bn_elu_out(float* __restrict__ out, const float* __restrict__ mu_inv,
                             const float* __restrict__ g, const float* __restrict__ b, long total) {
  long stride = (long)gridDim.x * blockDim.x;
  for (long i = blockIdx.x * (long)blockDim.x + threadIdx.x; i < total; i += stride) {
    int c = (int)(i & 63);
    float v = out[i];
    v = g[c] * (v - mu_inv[c]) * mu_inv[CC + c] + b[c];
    out[i] = elu1(v);
  }
}

// ---------------- launcher ----------------

extern "C" void kernel_launch(void* const* d_in, const int* in_sizes, int n_in,
                              void* d_out, int out_size, void* d_ws, size_t ws_size,
                              hipStream_t stream) {
  const float* x      = (const float*)d_in[0];
  const int*   ei     = (const int*)d_in[1];
  const float* pos    = (const float*)d_in[2];
  const float* yfeat  = (const float*)d_in[3];
  const float* W_lin  = (const float*)d_in[4];
  const float* W_lift = (const float*)d_in[5];
  const float* b_lift = (const float*)d_in[6];
  const float* g0     = (const float*)d_in[7];
  const float* b0     = (const float*)d_in[8];
  const float* W_hid  = (const float*)d_in[9];
  const float* b_hid  = (const float*)d_in[10];
  const float* g1     = (const float*)d_in[11];
  const float* b1     = (const float*)d_in[12];
  const float* W_low  = (const float*)d_in[13];
  const float* b_low  = (const float*)d_in[14];
  const float* bn_g   = (const float*)d_in[15];
  const float* bn_b   = (const float*)d_in[16];

  int  N  = in_sizes[0] / CC;
  long E  = in_sizes[1] / 2;
  long E2 = E + N;
  int ntE = (int)((E2 + 15) / 16);
  int ntN = (N + 15) / 16;

  // workspace carve-up
  char* ws = (char*)d_ws;
  size_t hbytes = ((size_t)ntE * 16 * CC * sizeof(_Float16) + 255) & ~(size_t)255;
  _Float16* h   = (_Float16*)ws;
  float* deg    = (float*)(ws + hbytes);
  float* agg    = deg + N;
  float* stats  = agg + (size_t)N * CC;       // 6*64: [s0 sum|sq][s1][s2]
  float* mu_inv = stats + 6 * CC;             // 3*128: [mu|inv] per BN
  (void)ws_size; (void)n_in; (void)out_size;

  float* pre = (float*)d_out;

  k_init<<<(N * CC + 255) / 256, 256, 0, stream>>>(deg, agg, stats, N);
  k_degree<<<(int)((E + 255) / 256), 256, 0, stream>>>(ei, deg, (int)E);

  k_edge_lift<<<1024, TPB, 0, stream>>>(pos, yfeat, ei, W_lift, b_lift, h,
                                        stats, E, E2, ntE);
  k_stats_fin<<<1, 64, 0, stream>>>(stats, mu_inv, (float)E2);

  // BN0+ELU fused into staging of the hidden GEMM (saves a full h round-trip)
  k_edge_hid<<<1024, TPB, 0, stream>>>(h, W_hid, b_hid, mu_inv, g0, b0,
                                       stats + 2 * CC, E2, ntE);
  k_stats_fin<<<1, 64, 0, stream>>>(stats + 2 * CC, mu_inv + 128, (float)E2);

  // BN1+ELU fused into staging of the low GEMM + scatter
  k_edge_low_scatter<<<1024, TPB, 0, stream>>>(h, W_low, b_low, mu_inv + 128, g1, b1,
                                               ei, x, deg, agg, E, E2, ntE);

  k_node<<<512, TPB, 0, stream>>>(x, W_lin, agg, pre, stats + 4 * CC, N, ntN);
  k_stats_fin<<<1, 64, 0, stream>>>(stats + 4 * CC, mu_inv + 256, (float)N);
  k_bn_elu_out<<<1024, 256, 0, stream>>>(pre, mu_inv + 256, bn_g, bn_b, (long)N * CC);
}
